// VMFLoss_7404523618945
// MI455X (gfx1250) — compile-verified, matrix operand order checked by reading, not run on hardware
//
#include <hip/hip_runtime.h>
#include <stdint.h>

#define NPTS 8192
#define DIM  512
#define BT   128         // block tile (M and N)
#define BK   64          // K chunk held in LDS
#define STR  72          // padded LDS row stride in halves (144B -> conflict-free b128 reads)
#define NCH  128         // column chunks per row in top3 scratch (64 cols each)
#define EPSC 1e-6f

typedef __attribute__((ext_vector_type(16))) _Float16 v16h;
typedef __attribute__((ext_vector_type(8)))  _Float16 v8h;
typedef __attribute__((ext_vector_type(8)))  float    v8f;

// ---------------- helpers ----------------

// keep (t1<=t2<=t3) = three smallest seen so far; insert v
__device__ __forceinline__ void insert3(float& t1, float& t2, float& t3, float v) {
  if (v < t3) {
    if (v < t2) {
      t3 = t2;
      if (v < t1) { t2 = t1; t1 = v; } else { t2 = v; }
    } else {
      t3 = v;
    }
  }
}

// merge two sorted triples -> three smallest of the union (into t1..t3)
__device__ __forceinline__ void merge3(float& t1, float& t2, float& t3,
                                       float b1, float b2, float b3) {
  float m1 = fminf(t1, b1), M1 = fmaxf(t1, b1);
  float m2 = fminf(t2, b2), M2 = fmaxf(t2, b2);
  float m3 = fminf(t3, b3);
  t1 = m1;
  t2 = fminf(M1, m2);
  t3 = fminf(fminf(fmaxf(M1, m2), M2), m3);
}

// Load a 16x32 f16 fragment (A layout; B uses the identical gather since B = X^T cols).
// lane<16 : row=lane,    halves [kk .. kk+7]  and [kk+16 .. kk+23]
// lane>=16: row=lane-16, halves [kk+8..kk+15] and [kk+24 .. kk+31]
__device__ __forceinline__ v16h ldsfrag(const _Float16* base, int kk, int lane) {
  union { v16h v; v8h h[2]; } u;
  const _Float16* p = base + (lane & 15) * STR + kk + ((lane >> 4) << 3);
  u.h[0] = *(const v8h*)(p);
  u.h[1] = *(const v8h*)(p + 16);
  return u.v;
}

// ---------------- kernels ----------------

__global__ void vmf_init_kernel(double* dacc, float* facc) {
  if (threadIdx.x == 0) {
    dacc[0] = 0.0;  // sum of 1/(dist^2+eps), off-diagonal
    dacc[1] = 0.0;  // sum of (d-1)*log(eps_i)
    facc[0] = 0.f;  // sum of ||x_i||
    facc[1] = 0.f;  // sum of ||x_i||^2
  }
}

__global__ __launch_bounds__(256) void vmf_prep_kernel(
    const float* __restrict__ x, _Float16* __restrict__ xh,
    float* __restrict__ sq, float* __restrict__ facc) {
  __shared__ float red[256];
  const int r = blockIdx.x;
  const int t = threadIdx.x;
  const float* row = x + (size_t)r * DIM;
  float a0 = row[t];
  float a1 = row[t + 256];
  red[t] = a0 * a0 + a1 * a1;
  __syncthreads();
  for (int s = 128; s > 0; s >>= 1) {
    if (t < s) red[t] += red[t + s];
    __syncthreads();
  }
  float sumsq = red[0];
  float norm  = sqrtf(sumsq);
  float inv   = 1.f / fmaxf(norm, 1e-12f);
  _Float16* orow = xh + (size_t)r * DIM;
  orow[t]       = (_Float16)(a0 * inv);
  orow[t + 256] = (_Float16)(a1 * inv);
  if (t == 0) {
    sq[r] = sumsq * inv * inv;   // ||x_unit||^2 (~1.0)
    atomicAdd(&facc[0], norm);
    atomicAdd(&facc[1], norm * norm);
  }
}

__global__ __launch_bounds__(256) void vmf_gram_kernel(
    const _Float16* __restrict__ xh, const float* __restrict__ sq,
    float* __restrict__ top3, double* __restrict__ dacc) {
  __shared__ _Float16 As[BT * STR];
  __shared__ _Float16 Bs[BT * STR];
  __shared__ float sqI[BT], sqJ[BT];
  __shared__ float red[256];

  const int jb = blockIdx.x, ib = blockIdx.y;
  const int I0 = ib * BT, J0 = jb * BT;
  const int t = threadIdx.x;
  const int w = t >> 5;
  const int lane = t & 31;
  // 2x4 wave tiling: wave owns 32 rows x 64 cols of the 128x128 block
  const int wr = w >> 1;           // 0..3 -> row slice
  const int wc = w & 1;            // 0..1 -> col slice
  const int R0 = wr << 5;          // local row base (32 rows)
  const int C0 = wc << 6;          // local col base (64 cols)

  if (t < BT) { sqI[t] = sq[I0 + t]; sqJ[t] = sq[J0 + t]; }

  v8f acc[2][4] = {};

  for (int kc = 0; kc < DIM; kc += BK) {
    // stage 128x64 halves of A and B panels into LDS (16B per thread x4 per panel)
    #pragma unroll
    for (int rpt = 0; rpt < 4; ++rpt) {
      int idx = t + rpt * 256;
      int row = idx >> 3, seg = idx & 7;
      *(uint4*)(As + row * STR + seg * 8) =
          *(const uint4*)(xh + (size_t)(I0 + row) * DIM + kc + seg * 8);
      *(uint4*)(Bs + row * STR + seg * 8) =
          *(const uint4*)(xh + (size_t)(J0 + row) * DIM + kc + seg * 8);
    }
    __syncthreads();

    #pragma unroll
    for (int kk = 0; kk < BK; kk += 32) {
      v16h a0 = ldsfrag(As + R0 * STR, kk, lane);
      v16h a1 = ldsfrag(As + (R0 + 16) * STR, kk, lane);
      #pragma unroll
      for (int nb = 0; nb < 4; ++nb) {
        v16h b = ldsfrag(Bs + (C0 + (nb << 4)) * STR, kk, lane);
        acc[0][nb] = __builtin_amdgcn_wmma_f32_16x16x32_f16(
            false, a0, false, b, (short)0, acc[0][nb], false, false);
        acc[1][nb] = __builtin_amdgcn_wmma_f32_16x16x32_f16(
            false, a1, false, b, (short)0, acc[1][nb], false, false);
      }
    }
    __syncthreads();
  }

  // ---- epilogue: C layout: VGPR e, lanes0-15 -> M=e, lanes16-31 -> M=e+8; N=lane&15
  const int half = lane >> 4;
  const int nn   = lane & 15;
  float invpart = 0.f;

  #pragma unroll
  for (int rb = 0; rb < 2; ++rb) {
    #pragma unroll
    for (int e = 0; e < 8; ++e) {
      const int li = R0 + (rb << 4) + e + (half << 3);  // local row in [0,128)
      const int gi = I0 + li;
      const float si = sqI[li];
      float t1 = 1e30f, t2 = 1e30f, t3 = 1e30f;
      #pragma unroll
      for (int nb = 0; nb < 4; ++nb) {
        const int lj = C0 + (nb << 4) + nn;
        const int gj = J0 + lj;
        float g = acc[rb][nb][e];
        if (gi != gj) {
          float cosv  = fminf(fmaxf(g, -1.f + EPSC), 1.f - EPSC);
          float chord = 2.f - 2.f * cosv;
          insert3(t1, t2, t3, chord);
          float dsq = si + sqJ[lj] - 2.f * g + 1e-6f;
          invpart += 1.f / (dsq + 1e-6f);
        }
      }
      // reduce three-smallest across the 16 lanes that share this row
      #pragma unroll
      for (int m = 1; m <= 8; m <<= 1) {
        float b1 = __shfl_xor(t1, m, 16);
        float b2 = __shfl_xor(t2, m, 16);
        float b3 = __shfl_xor(t3, m, 16);
        merge3(t1, t2, t3, b1, b2, b3);
      }
      if (nn == 0) {
        // column chunk id: 2 per jb block (wc selects which 64-col half)
        float* dst = top3 + ((size_t)gi * NCH + (jb << 1) + wc) * 3;
        dst[0] = t1; dst[1] = t2; dst[2] = t3;
      }
    }
  }

  // block-reduce the repulsion partial, one f64 atomic per block
  red[t] = invpart;
  __syncthreads();
  for (int s = 128; s > 0; s >>= 1) {
    if (t < s) red[t] += red[t + s];
    __syncthreads();
  }
  if (t == 0) atomicAdd(&dacc[0], (double)red[0]);
}

__global__ __launch_bounds__(256) void vmf_knn_merge_kernel(
    const float* __restrict__ top3, double* __restrict__ dacc) {
  __shared__ float red[256];
  const int r = blockIdx.x * 256 + threadIdx.x;
  const float* p = top3 + (size_t)r * (NCH * 3);
  float t1 = 1e30f, t2 = 1e30f, t3 = 1e30f;
  for (int i = 0; i < NCH * 3; i += 3) {
    merge3(t1, t2, t3, p[i], p[i + 1], p[i + 2]);
  }
  float eps_i = t3 + EPSC;                      // 3rd-smallest chord^2 + EPS
  red[threadIdx.x] = (float)(DIM - 1) * logf(eps_i);
  __syncthreads();
  for (int s = 128; s > 0; s >>= 1) {
    if (threadIdx.x < s) red[threadIdx.x] += red[threadIdx.x + s];
    __syncthreads();
  }
  if (threadIdx.x == 0) atomicAdd(&dacc[1], (double)red[0]);
}

__global__ void vmf_finalize_kernel(const double* __restrict__ dacc,
                                    const float* __restrict__ facc,
                                    const float* __restrict__ rs,
                                    float* __restrict__ out) {
  if (threadIdx.x == 0 && blockIdx.x == 0) {
    const double n = (double)NPTS;
    double h      = -(dacc[1] / n);
    double meann  = (double)facc[0] / n;
    double radius = (double)facc[1] / n - meann * meann;
    double invm   = dacc[0] / (n * (n - 1.0));
    float  sg     = 1.f / (1.f + expf(-rs[0]));
    double rep    = 0.2 * (double)sg * invm;
    out[0] = (float)(h + radius + rep);
    out[1] = (float)h;
    out[2] = (float)radius;
    out[3] = (float)rep;
  }
}

// ---------------- launcher ----------------

extern "C" void kernel_launch(void* const* d_in, const int* in_sizes, int n_in,
                              void* d_out, int out_size, void* d_ws, size_t ws_size,
                              hipStream_t stream) {
  (void)in_sizes; (void)n_in; (void)out_size; (void)ws_size;
  const float* x  = (const float*)d_in[0];
  const float* rs = (const float*)d_in[1];
  float* out = (float*)d_out;

  uint8_t* ws = (uint8_t*)d_ws;
  double*   dacc = (double*)ws;                               // 2 doubles
  float*    facc = (float*)(ws + 16);                         // 2 floats
  float*    sq   = (float*)(ws + 64);                         // 8192 f32
  _Float16* xh   = (_Float16*)(ws + 64 + (size_t)NPTS * 4);   // 8192x512 f16 (8 MB)
  float*    top3 = (float*)(ws + 64 + (size_t)NPTS * 4
                               + (size_t)NPTS * DIM * 2);     // 8192x128x3 f32 (12 MB)

  vmf_init_kernel<<<1, 32, 0, stream>>>(dacc, facc);
  vmf_prep_kernel<<<NPTS, 256, 0, stream>>>(x, xh, sq, facc);
  dim3 grid(NPTS / BT, NPTS / BT);
  vmf_gram_kernel<<<grid, 256, 0, stream>>>(xh, sq, top3, dacc);
  vmf_knn_merge_kernel<<<NPTS / 256, 256, 0, stream>>>(top3, dacc);
  vmf_finalize_kernel<<<1, 32, 0, stream>>>(dacc, facc, rs, out);
}
